// Block_34394098106804
// MI455X (gfx1250) — matrix operand order repaired and checked by reading, not compiled, not run on hardware
//
#include <hip/hip_runtime.h>
#include <stddef.h>

// Problem constants (reference: B,T,C,H,E = 4,2048,768,12,8; DFF=4C; HD=64)
#define BD   4
#define TT   2048
#define CC   768
#define HH   12
#define HDIM 64
#define EE   8
#define DFFN 3072
#define BT   (BD*TT)                       // 8192 token rows
#define PLANE ((size_t)BD*HH*TT*HDIM)      // elems per q/k/v plane: 6,291,456

typedef __attribute__((ext_vector_type(16))) __bf16 v16bf;
typedef __attribute__((ext_vector_type(8)))  float  v8f;
typedef __attribute__((ext_vector_type(4)))  unsigned int v4u;
typedef __attribute__((ext_vector_type(4)))  unsigned int v4ui;
typedef __attribute__((ext_vector_type(8)))  int v8i;
typedef __attribute__((ext_vector_type(4)))  int v4i;

union FragBF { v16bf bf; v4u u[2]; };

__device__ __forceinline__ unsigned short f2bf(float f) {
  unsigned u = __float_as_uint(f);
  u += 0x7FFFu + ((u >> 16) & 1u);          // round-to-nearest-even
  return (unsigned short)(u >> 16);
}
__device__ __forceinline__ float bf2f(unsigned short h) {
  return __uint_as_float(((unsigned)h) << 16);
}
__device__ __forceinline__ v8f wmma_bf16(v16bf a, v16bf b, v8f c) {
  return __builtin_amdgcn_wmma_f32_16x16x32_bf16(false, a, false, b, (short)0, c,
                                                 false, false);
}

// ---------------------------------------------------------------------------
// Tensor Data Mover: async-load one 128-row x 32-elem bf16 tile (row-major,
// row stride = `stride0` elems) into LDS at byte offset `lds_off`, padding
// each 64B row with 16B so LDS row stride is 80B (matches ushort[...][40]).
// D# layout per CDNA5 ISA. Tracked on TENSORcnt.
// ---------------------------------------------------------------------------
__device__ __forceinline__ void tdm_tile_128x32(unsigned lds_off,
                                                const unsigned short* g,
                                                unsigned dim0, unsigned dim1,
                                                unsigned stride0) {
  const unsigned long long ga = (unsigned long long)(size_t)g;
  v4ui g0;
  g0[0] = 1u;                                          // count=1, user mode
  g0[1] = lds_off;                                     // lds_addr
  g0[2] = (unsigned)ga;                                // global_addr[31:0]
  g0[3] = ((unsigned)(ga >> 32) & 0x01FFFFFFu) | (2u << 30);  // addr[56:32], type=2
  v8i g1;
  g1[0] = (int)((1u << 16)      // data_size = 2B
              | (1u << 20)      // pad_enable
              | (3u << 22)      // pad_interval: 16 DWORDs (64B)
              | (3u << 25));    // pad_amount: 4 DWORDs (16B)
  g1[1] = (int)((dim0 & 0xFFFFu) << 16);               // tensor_dim0[15:0]
  g1[2] = (int)((dim0 >> 16) | ((dim1 & 0xFFFFu) << 16));  // dim0[31:16], dim1[15:0]
  g1[3] = (int)((dim1 >> 16) | (32u << 16));           // dim1[31:16], tile_dim0=32
  g1[4] = (int)128;                                    // tile_dim1=128, tile_dim2=0
  g1[5] = (int)stride0;                                // tensor_dim0_stride[31:0]
  g1[6] = 0;                                           // stride0[47:32], stride1 lo
  g1[7] = 0;
  const v4i z4 = {0, 0, 0, 0};
#if defined(__clang_major__) && (__clang_major__ >= 23)
  const v8i z8 = {0, 0, 0, 0, 0, 0, 0, 0};
  __builtin_amdgcn_tensor_load_to_lds(g0, g1, z4, z4, z8, 0);
#else
  __builtin_amdgcn_tensor_load_to_lds(g0, g1, z4, z4, 0);
#endif
}

// ---------------------------------------------------------------------------
// Weight prep: W[K][N] f32 (batched over blockIdx.z) -> W^T[N][K] bf16.
// ---------------------------------------------------------------------------
__global__ __launch_bounds__(256) void wtrans_kernel(const float* __restrict__ src,
                                                     unsigned short* __restrict__ dst,
                                                     int K, int N) {
  __shared__ unsigned short t[32][33];
  const size_t mat = (size_t)blockIdx.z * (size_t)K * N;
  const int nb = blockIdx.x * 32, kb = blockIdx.y * 32;
  const int c = threadIdx.x & 31, r0 = threadIdx.x >> 5;
  const float* s = src + mat + (size_t)kb * N + nb;
#pragma unroll
  for (int i = 0; i < 4; ++i) {
    const int r = r0 + i * 8;
    t[c][r] = f2bf(s[(size_t)r * N + c]);
  }
  __syncthreads();
  unsigned short* d = dst + mat + (size_t)nb * K + kb;
#pragma unroll
  for (int i = 0; i < 4; ++i) {
    const int r = r0 + i * 8;
    d[(size_t)r * K + c] = t[r][c];
  }
}

// ---------------------------------------------------------------------------
// LayerNorm over C=768, f32 in -> bf16 out. One 256-thread block per row.
// ---------------------------------------------------------------------------
__global__ __launch_bounds__(256) void ln_kernel(const float* __restrict__ x,
                                                 const float* __restrict__ w,
                                                 const float* __restrict__ b,
                                                 unsigned short* __restrict__ out) {
  const int row = blockIdx.x;
  const int tid = threadIdx.x;
  const float* xr = x + (size_t)row * CC;
  float v0 = xr[tid], v1 = xr[tid + 256], v2 = xr[tid + 512];
  __shared__ float red[8];
  float s = v0 + v1 + v2;
#pragma unroll
  for (int m = 16; m >= 1; m >>= 1) s += __shfl_xor(s, m, 32);
  if ((tid & 31) == 0) red[tid >> 5] = s;
  __syncthreads();
  float tot = 0.f;
#pragma unroll
  for (int i = 0; i < 8; ++i) tot += red[i];
  const float mean = tot * (1.0f / CC);
  const float d0 = v0 - mean, d1 = v1 - mean, d2 = v2 - mean;
  s = d0 * d0 + d1 * d1 + d2 * d2;
  __syncthreads();
#pragma unroll
  for (int m = 16; m >= 1; m >>= 1) s += __shfl_xor(s, m, 32);
  if ((tid & 31) == 0) red[tid >> 5] = s;
  __syncthreads();
  tot = 0.f;
#pragma unroll
  for (int i = 0; i < 8; ++i) tot += red[i];
  const float rstd = rsqrtf(tot * (1.0f / CC) + 1e-5f);
  unsigned short* orow = out + (size_t)row * CC;
  orow[tid]       = f2bf(d0 * rstd * w[tid]       + b[tid]);
  orow[tid + 256] = f2bf(d1 * rstd * w[tid + 256] + b[tid + 256]);
  orow[tid + 512] = f2bf(d2 * rstd * w[tid + 512] + b[tid + 512]);
}

// ---------------------------------------------------------------------------
// Gate: softmax(h2 @ gate_w + gate_b) over E=8. One wave per token.
// ---------------------------------------------------------------------------
__global__ __launch_bounds__(256) void gate_kernel(const unsigned short* __restrict__ h,
                                                   const float* __restrict__ gw,
                                                   const float* __restrict__ gb,
                                                   float* __restrict__ gate) {
  const int token = blockIdx.x * 8 + (threadIdx.x >> 5);
  const int lane = threadIdx.x & 31;
  float acc[EE];
#pragma unroll
  for (int e = 0; e < EE; ++e) acc[e] = 0.f;
  const unsigned short* hr = h + (size_t)token * CC;
  for (int c = lane; c < CC; c += 32) {
    const float hv = bf2f(hr[c]);
    const float* gr = gw + (size_t)c * EE;
#pragma unroll
    for (int e = 0; e < EE; ++e) acc[e] += hv * gr[e];
  }
#pragma unroll
  for (int e = 0; e < EE; ++e) {
#pragma unroll
    for (int m = 16; m >= 1; m >>= 1) acc[e] += __shfl_xor(acc[e], m, 32);
    acc[e] += gb[e];
  }
  float mx = acc[0];
#pragma unroll
  for (int e = 1; e < EE; ++e) mx = fmaxf(mx, acc[e]);
  float sum = 0.f;
#pragma unroll
  for (int e = 0; e < EE; ++e) { acc[e] = __expf(acc[e] - mx); sum += acc[e]; }
  const float inv = 1.0f / sum;
  if (lane == 0) {
#pragma unroll
    for (int e = 0; e < EE; ++e) gate[(size_t)token * EE + e] = acc[e] * inv;
  }
}

// ---------------------------------------------------------------------------
// bf16 WMMA GEMM: C[M,N] = A[M,K] @ Bt[N,K]^T, both bf16 row-major.
// TDM double-buffered LDS staging: wave 0 issues tensor_load_to_lds for the
// next 128x32 A and B tiles while all 8 waves run WMMA on the current buffer.
//   EPI 0: QKV split   -> bf16 [3][B,H,T,64]
//   EPI 1: proj + x    -> f32 x1
//   EPI 2: bias+GELU   -> bf16 hid
//   EPI 3: out += gate[:,e] * (acc + bias)
// Block: 256 threads = 8 waves (4x2 wave grid); tile 128x128x32.
// ---------------------------------------------------------------------------
template <int EPI>
__global__ __launch_bounds__(256) void gemm_bf16_kern(
    const unsigned short* __restrict__ A, const unsigned short* __restrict__ Bt,
    const float* __restrict__ bias, int M, int N, int K,
    float* __restrict__ outf, unsigned short* __restrict__ outh,
    const float* __restrict__ resid, const float* __restrict__ gatep, int expert) {
  __shared__ __align__(16) unsigned short As[2][128][40];  // 80B row stride (TDM pad)
  __shared__ __align__(16) unsigned short Bs[2][128][40];
  const int n0 = blockIdx.x * 128, m0 = blockIdx.y * 128;
  const int tid = threadIdx.x;
  const int wave = tid >> 5, lane = tid & 31, half = lane >> 4, l16 = lane & 15;
  const int wm = wave >> 1, wn = wave & 1;
  v8f acc[2][4] = {};

  const unsigned ldsA[2] = {(unsigned)(size_t)(void*)&As[0][0][0],
                            (unsigned)(size_t)(void*)&As[1][0][0]};
  const unsigned ldsB[2] = {(unsigned)(size_t)(void*)&Bs[0][0][0],
                            (unsigned)(size_t)(void*)&Bs[1][0][0]};
  const unsigned short* Abase = A + (size_t)m0 * K;
  const unsigned short* Bbase = Bt + (size_t)n0 * K;
  const int nk = K >> 5;

  if (tid < 32) {  // wave 0: kick off first tile pair
    tdm_tile_128x32(ldsA[0], Abase, (unsigned)K, (unsigned)M, (unsigned)K);
    tdm_tile_128x32(ldsB[0], Bbase, (unsigned)K, (unsigned)N, (unsigned)K);
  }

  for (int it = 0; it < nk; ++it) {
    const int cur = it & 1;
    if (tid < 32) {
      if (it + 1 < nk) {  // prefetch next tiles into the other buffer
        const int kt = (it + 1) << 5;
        tdm_tile_128x32(ldsA[cur ^ 1], Abase + kt, (unsigned)K, (unsigned)M,
                        (unsigned)K);
        tdm_tile_128x32(ldsB[cur ^ 1], Bbase + kt, (unsigned)K, (unsigned)N,
                        (unsigned)K);
        __builtin_amdgcn_s_wait_tensorcnt(2);  // current pair complete
      } else {
        __builtin_amdgcn_s_wait_tensorcnt(0);
      }
    }
    __syncthreads();  // publish current tiles to all waves

    FragBF a[2], bfr[4];
#pragma unroll
    for (int mi = 0; mi < 2; ++mi) {  // A-frag: K {0-7,16-23} / {8-15,24-31}
      const int r = wm * 32 + mi * 16 + l16;
      a[mi].u[0] = *(const v4u*)&As[cur][r][half * 8];
      a[mi].u[1] = *(const v4u*)&As[cur][r][16 + half * 8];
    }
#pragma unroll
    for (int ni = 0; ni < 4; ++ni) {  // B-frag: K {0-7,16-23} / {8-15,24-31}
      const int n = wn * 64 + ni * 16 + l16;
      bfr[ni].u[0] = *(const v4u*)&Bs[cur][n][half * 8];
      bfr[ni].u[1] = *(const v4u*)&Bs[cur][n][16 + half * 8];
    }
#pragma unroll
    for (int mi = 0; mi < 2; ++mi)
#pragma unroll
      for (int ni = 0; ni < 4; ++ni)
        acc[mi][ni] = wmma_bf16(a[mi].bf, bfr[ni].bf, acc[mi][ni]);
    __syncthreads();  // current buffer free for the next TDM write
  }

  // Epilogue: C layout — VGPR v: rows v / v+8 (per half), lane&15 = col.
#pragma unroll
  for (int mi = 0; mi < 2; ++mi) {
#pragma unroll
    for (int ni = 0; ni < 4; ++ni) {
#pragma unroll
      for (int v = 0; v < 8; ++v) {
        const int row = m0 + wm * 32 + mi * 16 + v + 8 * half;
        const int col = n0 + wn * 64 + ni * 16 + l16;
        const float val = acc[mi][ni][v] + bias[col];
        if (EPI == 0) {
          const int which = col / CC, r = col - which * CC;
          const int hh = r >> 6, hd = r & 63;
          const int bq = row >> 11, t = row & (TT - 1);
          outh[(size_t)which * PLANE +
               (((size_t)(bq * HH + hh) * TT + t) << 6) + hd] = f2bf(val);
        } else if (EPI == 1) {
          const size_t idx = (size_t)row * CC + col;
          outf[idx] = resid[idx] + val;
        } else if (EPI == 2) {
          const float g = 0.5f * val * (1.0f + erff(val * 0.70710678118654752f));
          outh[(size_t)row * DFFN + col] = f2bf(g);
        } else {
          const size_t idx = (size_t)row * CC + col;
          outf[idx] += gatep[(size_t)row * EE + expert] * val;
        }
      }
    }
  }
}

// ---------------------------------------------------------------------------
// Flash attention, causal. Grid: (T/64 q-blocks, B*H). Block: 128 thr = 4 waves.
// ---------------------------------------------------------------------------
__global__ __launch_bounds__(128) void attn_kernel(const unsigned short* __restrict__ qkv,
                                                   unsigned short* __restrict__ y) {
  __shared__ __align__(16) unsigned short Ks[64][72];       // [token][hd]
  __shared__ __align__(16) unsigned short Vs[64][72];       // [hd][token]
  __shared__ __align__(16) unsigned short Ps[4][16][72];    // per-wave P tile
  const int bh = blockIdx.y;
  const int bb = bh / HH, hh = bh % HH;
  const int q0 = blockIdx.x * 64;
  const int tid = threadIdx.x;
  const int wave = tid >> 5, lane = tid & 31, half = lane >> 4, l16 = lane & 15;
  const unsigned short* Q  = qkv + (size_t)bh * TT * HDIM;
  const unsigned short* Kp = qkv + PLANE + (size_t)bh * TT * HDIM;
  const unsigned short* Vp = qkv + 2 * PLANE + (size_t)bh * TT * HDIM;

  FragBF aQ[2];
  {
    const int qr = q0 + wave * 16 + l16;
#pragma unroll
    for (int ks = 0; ks < 2; ++ks) {
      const unsigned short* p = Q + (size_t)qr * HDIM + ks * 32;
      aQ[ks].u[0] = *(const v4u*)(p + half * 8);
      aQ[ks].u[1] = *(const v4u*)(p + 16 + half * 8);
    }
  }
  v8f accO[4] = {};
  float mrow[8], lrow[8];
#pragma unroll
  for (int v = 0; v < 8; ++v) { mrow[v] = -1e30f; lrow[v] = 0.f; }

  const int jend = blockIdx.x;
  for (int j = 0; j <= jend; ++j) {
    const int k0 = j * 64;
    {  // stage K (row-major) and V (transposed); prefetch next tiles
      const int tok = tid >> 1, seg = (tid & 1) * 32;
      const v4u* ksrc = (const v4u*)(Kp + (size_t)(k0 + tok) * HDIM + seg);
      const v4u k0v = ksrc[0], k1v = ksrc[1], k2v = ksrc[2], k3v = ksrc[3];
      *(v4u*)&Ks[tok][seg]      = k0v;
      *(v4u*)&Ks[tok][seg + 8]  = k1v;
      *(v4u*)&Ks[tok][seg + 16] = k2v;
      *(v4u*)&Ks[tok][seg + 24] = k3v;
      union { v4u u[4]; unsigned short s[32]; } tv;
      const v4u* vsrc = (const v4u*)(Vp + (size_t)(k0 + tok) * HDIM + seg);
      tv.u[0] = vsrc[0]; tv.u[1] = vsrc[1]; tv.u[2] = vsrc[2]; tv.u[3] = vsrc[3];
#pragma unroll
      for (int i = 0; i < 32; ++i) Vs[seg + i][tok] = tv.s[i];
      if (j < jend) {  // global_prefetch_b8 of the next K/V tiles
        __builtin_prefetch(Kp + (size_t)(k0 + 64 + tok) * HDIM + seg, 0, 0);
        __builtin_prefetch(Vp + (size_t)(k0 + 64 + tok) * HDIM + seg, 0, 0);
      }
    }
    __syncthreads();

    v8f accS[4] = {};
#pragma unroll
    for (int ks = 0; ks < 2; ++ks) {
#pragma unroll
      for (int ni = 0; ni < 4; ++ni) {
        FragBF bK;
        const int n = ni * 16 + l16;
        const int o = ks * 32 + half * 8;
        bK.u[0] = *(const v4u*)&Ks[n][o];
        bK.u[1] = *(const v4u*)&Ks[n][o + 16];
        accS[ni] = wmma_bf16(aQ[ks].bf, bK.bf, accS[ni]);
      }
    }
#pragma unroll
    for (int ni = 0; ni < 4; ++ni) {
      const int col = k0 + ni * 16 + l16;
#pragma unroll
      for (int v = 0; v < 8; ++v) {
        const int row = q0 + wave * 16 + v + 8 * half;
        const float s = accS[ni][v] * 0.125f;  // 1/sqrt(64)
        accS[ni][v] = (col > row) ? -1e30f : s;
      }
    }
    float corr[8], rsum[8];
#pragma unroll
    for (int v = 0; v < 8; ++v) {
      float x = fmaxf(fmaxf(accS[0][v], accS[1][v]), fmaxf(accS[2][v], accS[3][v]));
      x = fmaxf(x, __shfl_xor(x, 1, 32));
      x = fmaxf(x, __shfl_xor(x, 2, 32));
      x = fmaxf(x, __shfl_xor(x, 4, 32));
      x = fmaxf(x, __shfl_xor(x, 8, 32));
      const float nm = fmaxf(mrow[v], x);
      corr[v] = __expf(mrow[v] - nm);
      mrow[v] = nm;
      rsum[v] = 0.f;
    }
#pragma unroll
    for (int ni = 0; ni < 4; ++ni)
#pragma unroll
      for (int v = 0; v < 8; ++v) {
        const float p = __expf(accS[ni][v] - mrow[v]);
        accS[ni][v] = p;
        rsum[v] += p;
      }
#pragma unroll
    for (int v = 0; v < 8; ++v) {
      float s = rsum[v];
      s += __shfl_xor(s, 1, 32);
      s += __shfl_xor(s, 2, 32);
      s += __shfl_xor(s, 4, 32);
      s += __shfl_xor(s, 8, 32);
      lrow[v] = lrow[v] * corr[v] + s;
#pragma unroll
      for (int ni = 0; ni < 4; ++ni) accO[ni][v] *= corr[v];
    }
#pragma unroll
    for (int ni = 0; ni < 4; ++ni)
#pragma unroll
      for (int v = 0; v < 8; ++v)
        Ps[wave][v + 8 * half][ni * 16 + l16] = f2bf(accS[ni][v]);
#pragma unroll
    for (int ks = 0; ks < 2; ++ks) {
      FragBF aP;
      aP.u[0] = *(const v4u*)&Ps[wave][l16][ks * 32 + half * 8];
      aP.u[1] = *(const v4u*)&Ps[wave][l16][ks * 32 + 16 + half * 8];
#pragma unroll
      for (int ni = 0; ni < 4; ++ni) {
        FragBF bV;
        const int n = ni * 16 + l16;
        const int o = ks * 32 + half * 8;
        bV.u[0] = *(const v4u*)&Vs[n][o];
        bV.u[1] = *(const v4u*)&Vs[n][o + 16];
        accO[ni] = wmma_bf16(aP.bf, bV.bf, accO[ni]);
      }
    }
    __syncthreads();
  }
#pragma unroll
  for (int ni = 0; ni < 4; ++ni) {
#pragma unroll
    for (int v = 0; v < 8; ++v) {
      const int row = q0 + wave * 16 + v + 8 * half;
      const int hd = ni * 16 + l16;
      const float val = accO[ni][v] / lrow[v];
      y[((size_t)(bb * TT + row)) * CC + hh * HDIM + hd] = f2bf(val);
    }
  }
}

// ---------------------------------------------------------------------------
extern "C" void kernel_launch(void* const* d_in, const int* in_sizes, int n_in,
                              void* d_out, int out_size, void* d_ws, size_t ws_size,
                              hipStream_t stream) {
  (void)in_sizes; (void)n_in; (void)out_size; (void)ws_size;
  const float* x      = (const float*)d_in[0];
  const float* ln1_w  = (const float*)d_in[1];
  const float* ln1_b  = (const float*)d_in[2];
  const float* attn_w = (const float*)d_in[3];
  const float* attn_b = (const float*)d_in[4];
  const float* proj_w = (const float*)d_in[5];
  const float* proj_b = (const float*)d_in[6];
  const float* ln2_w  = (const float*)d_in[7];
  const float* ln2_b  = (const float*)d_in[8];
  const float* gate_w = (const float*)d_in[9];
  const float* gate_b = (const float*)d_in[10];
  const float* fc_w   = (const float*)d_in[11];
  const float* fc_b   = (const float*)d_in[12];
  const float* pr_w   = (const float*)d_in[13];
  const float* pr_b   = (const float*)d_in[14];
  float* out = (float*)d_out;

  char* ws = (char*)d_ws;
  size_t off = 0;
  auto take = [&](size_t bytes) -> char* {
    char* p = ws + off;
    off = (off + bytes + 255) & ~(size_t)255;
    return p;
  };
  unsigned short* h1      = (unsigned short*)take((size_t)BT * CC * 2);
  unsigned short* qkv     = (unsigned short*)take((size_t)3 * PLANE * 2);
  unsigned short* yb      = (unsigned short*)take((size_t)BT * CC * 2);
  float*          x1      = (float*)take((size_t)BT * CC * 4);
  unsigned short* h2      = (unsigned short*)take((size_t)BT * CC * 2);
  float*          gate    = (float*)take((size_t)BT * EE * 4);
  unsigned short* hid     = (unsigned short*)take((size_t)BT * DFFN * 2);
  unsigned short* attn_wt = (unsigned short*)take((size_t)3 * CC * CC * 2);   // [3C][C]
  unsigned short* proj_wt = (unsigned short*)take((size_t)CC * CC * 2);       // [C][C]
  unsigned short* fc_wt   = (unsigned short*)take((size_t)EE * CC * DFFN * 2);// [E][DFF][C]
  unsigned short* pr_wt   = (unsigned short*)take((size_t)EE * CC * DFFN * 2);// [E][C][DFF]

  // 0) one-shot weight prep: f32 [K][N] -> bf16 [N][K]
  wtrans_kernel<<<dim3(3 * CC / 32, CC / 32, 1), 256, 0, stream>>>(attn_w, attn_wt, CC, 3 * CC);
  wtrans_kernel<<<dim3(CC / 32, CC / 32, 1), 256, 0, stream>>>(proj_w, proj_wt, CC, CC);
  wtrans_kernel<<<dim3(DFFN / 32, CC / 32, EE), 256, 0, stream>>>(fc_w, fc_wt, CC, DFFN);
  wtrans_kernel<<<dim3(CC / 32, DFFN / 32, EE), 256, 0, stream>>>(pr_w, pr_wt, DFFN, CC);

  // 1) LN1 -> bf16
  ln_kernel<<<BT, 256, 0, stream>>>(x, ln1_w, ln1_b, h1);
  // 2) QKV GEMM (M=8192, N=2304, K=768), split into [B,H,T,64] planes
  gemm_bf16_kern<0><<<dim3(3 * CC / 128, BT / 128), 256, 0, stream>>>(
      h1, attn_wt, attn_b, BT, 3 * CC, CC, nullptr, qkv, nullptr, nullptr, 0);
  // 3) causal flash attention -> y bf16
  attn_kernel<<<dim3(TT / 64, BD * HH), 128, 0, stream>>>(qkv, yb);
  // 4) proj GEMM + residual -> x1 (f32)
  gemm_bf16_kern<1><<<dim3(CC / 128, BT / 128), 256, 0, stream>>>(
      yb, proj_wt, proj_b, BT, CC, CC, x1, nullptr, x, nullptr, 0);
  // 5) LN2 -> bf16
  ln_kernel<<<BT, 256, 0, stream>>>(x1, ln2_w, ln2_b, h2);
  // 6) gate softmax
  gate_kernel<<<BT / 8, 256, 0, stream>>>(h2, gate_w, gate_b, gate);
  // 7) out = x1, then accumulate gated expert outputs
  hipMemcpyAsync(out, x1, (size_t)BT * CC * 4, hipMemcpyDeviceToDevice, stream);
  for (int e = 0; e < EE; ++e) {
    gemm_bf16_kern<2><<<dim3(DFFN / 128, BT / 128), 256, 0, stream>>>(
        h2, fc_wt + (size_t)e * CC * DFFN, fc_b + (size_t)e * DFFN,
        BT, DFFN, CC, nullptr, hid, nullptr, nullptr, e);
    gemm_bf16_kern<3><<<dim3(CC / 128, BT / 128), 256, 0, stream>>>(
        hid, pr_wt + (size_t)e * CC * DFFN, pr_b + (size_t)e * CC,
        BT, CC, DFFN, out, nullptr, nullptr, gate, e);
  }
}
